// MixtralDecoderLayer_35338990911708
// MI455X (gfx1250) — matrix operand order repaired and checked by reading, not compiled, not run on hardware
//
#include <hip/hip_runtime.h>
#include <hip/hip_bf16.h>

// Flip to 0 if the gfx1250 assembler rejects the async-to-LDS mnemonics.
#define USE_ASYNC_COPY 1

// ---------------- model constants ----------------
constexpr int   TT     = 2048;
constexpr int   HID_   = 2048;
constexpr int   FFN_   = 7168;
constexpr int   NH_    = 16;
constexpr int   NKV_   = 4;
constexpr int   HD_    = 128;
constexpr int   NE_    = 8;
constexpr float EPS_   = 1e-5f;
constexpr float THETA_ = 1000000.0f;
constexpr int   QKV_N  = NH_ * HD_ + 2 * NKV_ * HD_;   // 3072
constexpr int   PADCAP = 4608;  // 2T rows + 8 experts * 64-row padding, /64 aligned

// ---------------- WMMA plumbing ----------------
typedef __attribute__((ext_vector_type(16))) __bf16 v16bf;
typedef __attribute__((ext_vector_type(8)))  float  v8f;
union Frag { v16bf v; unsigned u[8]; };

__device__ inline unsigned short f2bf(float x) {          // RNE f32->bf16
  unsigned u = __float_as_uint(x);
  u += 0x7fffu + ((u >> 16) & 1u);
  return (unsigned short)(u >> 16);
}
__device__ inline float bf2f(unsigned short h) { return __uint_as_float(((unsigned)h) << 16); }
__device__ inline unsigned pack2(float a, float b) {
#if defined(__has_builtin)
#if __has_builtin(__builtin_amdgcn_cvt_pk_bf16_f32)
  auto r = __builtin_amdgcn_cvt_pk_bf16_f32(a, b);       // v_cvt_pk_bf16_f32
  return __builtin_bit_cast(unsigned, r);
#else
  return (unsigned)f2bf(a) | ((unsigned)f2bf(b) << 16);
#endif
#else
  return (unsigned)f2bf(a) | ((unsigned)f2bf(b) << 16);
#endif
}
__device__ inline v8f wmma_bf16(const Frag& a, const Frag& b, v8f c) {
  return __builtin_amdgcn_wmma_f32_16x16x32_bf16(false, a.v, false, b.v, (short)0, c, false, false);
}
__device__ inline void zero8(v8f& a) {
#pragma unroll
  for (int i = 0; i < 8; ++i) a[i] = 0.f;
}
__device__ inline float wave_sum(float s) {
#pragma unroll
  for (int m = 16; m >= 1; m >>= 1) s += __shfl_xor(s, m);
  return s;
}
__device__ inline float half_max(float s) {               // reduce across a 16-lane half
#pragma unroll
  for (int m = 8; m >= 1; m >>= 1) s = fmaxf(s, __shfl_xor(s, m));
  return s;
}
__device__ inline float half_sum(float s) {
#pragma unroll
  for (int m = 8; m >= 1; m >>= 1) s += __shfl_xor(s, m);
  return s;
}

#if USE_ASYNC_COPY
// gfx1250 async global->LDS DMA (ASYNCcnt tracked); inline asm since the clang
// builtin arity differs across toolchains. dst = 32-bit LDS byte address,
// src = 64-bit global address, 16 bytes per lane.
__device__ inline void async_copy_b128(unsigned lds_addr, unsigned long long gaddr) {
  asm volatile("global_load_async_to_lds_b128 %0, %1, off"
               :: "v"(lds_addr), "v"(gaddr) : "memory");
}
__device__ inline void async_wait0() {
  asm volatile("s_wait_asynccnt 0x0" ::: "memory");
}
__device__ inline unsigned lds_addr_of(const void* p) {   // LDS aperture: low 32 bits
  return (unsigned)(unsigned long long)p;
}
#endif

// ---------------- tiled bf16 WMMA GEMM core ----------------
// C_tile[BM x BN] = A[bf16, MxK row-major, optional row gather] @ B[f32, NxK row-major]^T
// 8 waves: 2(m) x 4(n) wave grid, each wave 32x32 via 2x2 v_wmma_f32_16x16x32_bf16.
// Double-buffered LDS; A staged by async DMA (bf16 already), B converted f32->bf16
// in-flight (weights read exactly once from HBM).
constexpr int BM = 64, BN = 128, BK = 32;

template <int NB>
__device__ inline void gemm_tile(const unsigned short* __restrict__ A,
                                 const float* __restrict__ B0,
                                 const float* __restrict__ B1,
                                 int K, int mbase, int nbase,
                                 const int* __restrict__ row_map,
                                 v8f acc[][2][2]) {
  __shared__ alignas(16) unsigned lds_a[2][BM][BK / 2];        // 8 KB
  __shared__ alignas(16) unsigned lds_b[2][NB][BN][BK / 2];    // 16/32 KB
  const int tid = threadIdx.x;
  const int lane = tid & 31, wid = tid >> 5;
  const int wm = wid >> 2, wn = wid & 3;
  const int half = lane >> 4, l16 = lane & 15;

  // staging assignments (vectorized, 16B granularity)
  const int ar = tid >> 2, ac = (tid & 3) * 4;        // A: 64 rows x 16 dw; 1 x b128/thread
  const int br = tid >> 1, bc = (tid & 1) * 16;       // B: 128 rows x 32 f32; 16 f32/thread
  const int aprow = row_map ? row_map[mbase + ar] : (mbase + ar);
  const unsigned short* aptr = A + (size_t)aprow * K + ac * 2;
  const float* bptr0 = B0 + (size_t)(nbase + br) * K + bc;
  const float* bptr1 = (NB > 1) ? (B1 + (size_t)(nbase + br) * K + bc) : nullptr;

  auto stage = [&](int buf, int kks) {
#if USE_ASYNC_COPY
    async_copy_b128(lds_addr_of(&lds_a[buf][ar][ac]),
                    (unsigned long long)(const void*)(aptr + kks));
#else
    *(uint4*)&lds_a[buf][ar][ac] = *(const uint4*)(aptr + kks);
#endif
#pragma unroll
    for (int nb = 0; nb < NB; ++nb) {
      const float* src = (nb ? bptr1 : bptr0) + kks;
      __builtin_prefetch(src + BK, 0, 1);               // speculative: next K tile
      float4 f0 = ((const float4*)src)[0];
      float4 f1 = ((const float4*)src)[1];
      float4 f2 = ((const float4*)src)[2];
      float4 f3 = ((const float4*)src)[3];
      uint4 p0, p1;
      p0.x = pack2(f0.x, f0.y); p0.y = pack2(f0.z, f0.w);
      p0.z = pack2(f1.x, f1.y); p0.w = pack2(f1.z, f1.w);
      p1.x = pack2(f2.x, f2.y); p1.y = pack2(f2.z, f2.w);
      p1.z = pack2(f3.x, f3.y); p1.w = pack2(f3.z, f3.w);
      uint4* dst = (uint4*)&lds_b[buf][nb][br][bc >> 1];
      dst[0] = p0;
      dst[1] = p1;
    }
  };

  stage(0, 0);
  int cur = 0;
  for (int kk = 0; kk < K; kk += BK) {
#if USE_ASYNC_COPY
    async_wait0();                                      // A DMA of 'cur' complete
#endif
    __syncthreads();                                    // tile 'cur' visible to all
    if (kk + BK < K) stage(cur ^ 1, kk + BK);           // overlap next tile with math

    Frag af[2];
#pragma unroll
    for (int i = 0; i < 2; ++i) {
      int r = wm * 32 + i * 16 + l16;
#pragma unroll
      for (int vv = 0; vv < 8; ++vv) {
        int kb = 2 * vv + (vv >= 4 ? 8 : 0) + half * 8; // ISA 16-bit A 16x32 layout
        af[i].u[vv] = lds_a[cur][r][kb >> 1];
      }
    }
#pragma unroll
    for (int nb = 0; nb < NB; ++nb) {
#pragma unroll
      for (int j = 0; j < 2; ++j) {
        Frag bfr;
        int n = wn * 32 + j * 16 + l16;
#pragma unroll
        for (int vv = 0; vv < 8; ++vv) {
          int k0 = 2 * vv + (vv >= 4 ? 8 : 0) + half * 8; // B: lane=N, VGPR holds K pair
          bfr.u[vv] = lds_b[cur][nb][n][k0 >> 1];
        }
#pragma unroll
        for (int i = 0; i < 2; ++i)
          acc[nb][i][j] = wmma_bf16(af[i], bfr, acc[nb][i][j]);
      }
    }
    cur ^= 1;
  }
}

// ---------------- RMSNorm (f32 in, bf16 out) ----------------
__global__ __launch_bounds__(256) void k_rmsnorm(const float* __restrict__ x,
                                                 const float* __restrict__ w,
                                                 unsigned short* __restrict__ out) {
  const int t = blockIdx.x;
  const float* xr = x + (size_t)t * HID_;
  float s = 0.f;
  for (int d = threadIdx.x; d < HID_; d += 256) { float v = xr[d]; s += v * v; }
  s = wave_sum(s);
  __shared__ float red[8];
  __shared__ float rinv;
  if ((threadIdx.x & 31) == 0) red[threadIdx.x >> 5] = s;
  __syncthreads();
  if (threadIdx.x == 0) {
    float tot = 0.f;
#pragma unroll
    for (int i = 0; i < 8; ++i) tot += red[i];
    rinv = rsqrtf(tot / HID_ + EPS_);
  }
  __syncthreads();
  const float r = rinv;
  unsigned short* orow = out + (size_t)t * HID_;
  for (int d = threadIdx.x; d < HID_; d += 256) orow[d] = f2bf(xr[d] * r * w[d]);
}

// ---------------- QKV projection ----------------
__global__ __launch_bounds__(256) void k_gemm_qkv(const unsigned short* __restrict__ A,
                                                  const float* __restrict__ B,
                                                  float* __restrict__ C) {
  const int nbase = blockIdx.x * BN, mbase = blockIdx.y * BM;
  v8f acc[1][2][2];
#pragma unroll
  for (int i = 0; i < 2; ++i)
#pragma unroll
    for (int j = 0; j < 2; ++j) zero8(acc[0][i][j]);
  gemm_tile<1>(A, B, nullptr, HID_, mbase, nbase, nullptr, acc);
  const int lane = threadIdx.x & 31, wid = threadIdx.x >> 5;
  const int wm = wid >> 2, wn = wid & 3, half = lane >> 4, l16 = lane & 15;
#pragma unroll
  for (int i = 0; i < 2; ++i)
#pragma unroll
    for (int j = 0; j < 2; ++j)
#pragma unroll
      for (int vi = 0; vi < 8; ++vi) {
        int row = mbase + wm * 32 + i * 16 + vi + half * 8;
        int col = nbase + wn * 32 + j * 16 + l16;
        C[(size_t)row * QKV_N + col] = acc[0][i][j][vi];
      }
}

// ---------------- O projection + residual ----------------
__global__ __launch_bounds__(256) void k_gemm_wo(const unsigned short* __restrict__ A,
                                                 const float* __restrict__ B,
                                                 const float* __restrict__ resid,
                                                 float* __restrict__ C) {
  const int nbase = blockIdx.x * BN, mbase = blockIdx.y * BM;
  v8f acc[1][2][2];
#pragma unroll
  for (int i = 0; i < 2; ++i)
#pragma unroll
    for (int j = 0; j < 2; ++j) zero8(acc[0][i][j]);
  gemm_tile<1>(A, B, nullptr, HID_, mbase, nbase, nullptr, acc);
  const int lane = threadIdx.x & 31, wid = threadIdx.x >> 5;
  const int wm = wid >> 2, wn = wid & 3, half = lane >> 4, l16 = lane & 15;
#pragma unroll
  for (int i = 0; i < 2; ++i)
#pragma unroll
    for (int j = 0; j < 2; ++j)
#pragma unroll
      for (int vi = 0; vi < 8; ++vi) {
        int row = mbase + wm * 32 + i * 16 + vi + half * 8;
        int col = nbase + wn * 32 + j * 16 + l16;
        size_t o = (size_t)row * HID_ + col;
        C[o] = resid[o] + acc[0][i][j][vi];
      }
}

// ---------------- RoPE + bf16 pack of q/k/v ----------------
__global__ __launch_bounds__(256) void k_rope(const float* __restrict__ qkv,
                                              const int* __restrict__ pos,
                                              unsigned short* __restrict__ q,
                                              unsigned short* __restrict__ k,
                                              unsigned short* __restrict__ v) {
  const int t = blockIdx.x;
  const float p = (float)pos[t];
  const float* row = qkv + (size_t)t * QKV_N;
  for (int idx = threadIdx.x; idx < (NH_ + NKV_) * 64; idx += 256) {
    const bool isq = idx < NH_ * 64;
    const int hi = isq ? idx : idx - NH_ * 64;
    const int h = hi >> 6, j = hi & 63;
    const float inv = __powf(THETA_, -(float)(2 * j) * (1.0f / HD_));
    float sn, cs;
    __sincosf(p * inv, &sn, &cs);
    const float* src = isq ? (row + h * HD_) : (row + NH_ * HD_ + h * HD_);
    const float x1 = src[j], x2 = src[j + 64];
    unsigned short* dst = isq ? (q + (size_t)t * NH_ * HD_ + h * HD_)
                              : (k + (size_t)t * NKV_ * HD_ + h * HD_);
    dst[j]      = f2bf(x1 * cs - x2 * sn);
    dst[j + 64] = f2bf(x2 * cs + x1 * sn);
  }
  for (int idx = threadIdx.x; idx < NKV_ * HD_; idx += 256)
    v[(size_t)t * NKV_ * HD_ + idx] = f2bf(row[NH_ * HD_ + NKV_ * HD_ + idx]);
}

// ---------------- flash attention ----------------
// 4 waves/block share one K/V LDS tile (4x K/V HBM reuse); each wave owns one
// 16-row q tile. QK^T and PV on WMMA; online softmax on 16-lane halves.
__global__ __launch_bounds__(128) void k_attn(const unsigned short* __restrict__ q,
                                              const unsigned short* __restrict__ k,
                                              const unsigned short* __restrict__ v,
                                              unsigned short* __restrict__ o) {
  const int h = blockIdx.x;
  const int kv = h >> 2;                                  // GQA: 4 q heads per kv head
  const int tid = threadIdx.x, wid = tid >> 5, lane = tid & 31;
  const int half = lane >> 4, l16 = lane & 15;
  const int qbase = blockIdx.y * 64 + wid * 16;           // this wave's q tile
  const int kend_wave  = qbase + 16;                      // causal bound (exclusive)
  const int kend_block = blockIdx.y * 64 + 64;

  __shared__ alignas(16) unsigned       k_lds[32][64];    // [key][d-pair]   8 KB
  __shared__ alignas(16) unsigned short v_lds[128][34];   // [d][key] padded 8.5 KB
  __shared__ alignas(16) unsigned short p_lds[4][16][32]; // per-wave probs  4 KB

  // persistent Q A-fragments for the 4 K=32 chunks of HD=128
  Frag af[4];
  {
    const unsigned* qp = (const unsigned*)(q + ((size_t)(qbase + l16) * NH_ + h) * HD_);
#pragma unroll
    for (int f = 0; f < 4; ++f)
#pragma unroll
      for (int vv = 0; vv < 8; ++vv) {
        int d = f * 32 + 2 * vv + (vv >= 4 ? 8 : 0) + half * 8;
        af[f].u[vv] = qp[d >> 1];
      }
  }
  v8f oacc[8];
#pragma unroll
  for (int c = 0; c < 8; ++c) zero8(oacc[c]);
  float mrow[8], lrow[8];
#pragma unroll
  for (int i = 0; i < 8; ++i) { mrow[i] = -1e30f; lrow[i] = 0.f; }
  const float scale = 0.08838834764831845f;               // HD^-0.5

  // staging assignment: 128 threads, 32 key rows, 4 threads/row
  const int sr = tid >> 2, sq = tid & 3;                  // row, quarter

  for (int kk = 0; kk < kend_block; kk += 32) {
    {                                                     // stage K (async DMA) + V (transpose)
      const unsigned short* krow = k + ((size_t)(kk + sr) * NKV_ + kv) * HD_;
      const unsigned short* vrow = v + ((size_t)(kk + sr) * NKV_ + kv) * HD_;
#if USE_ASYNC_COPY
#pragma unroll
      for (int c = 0; c < 4; ++c)
        async_copy_b128(lds_addr_of(&k_lds[sr][(sq * 4 + c) * 4]),
                        (unsigned long long)(const void*)(krow + (sq * 4 + c) * 8));
#else
#pragma unroll
      for (int c = 0; c < 4; ++c)
        ((uint4*)k_lds[sr])[sq * 4 + c] = ((const uint4*)krow)[sq * 4 + c];
#endif
#pragma unroll
      for (int c = 0; c < 4; ++c) {
        uint4 w = ((const uint4*)vrow)[sq * 4 + c];
        const int d0 = (sq * 4 + c) * 8;
        v_lds[d0 + 0][sr] = (unsigned short)(w.x); v_lds[d0 + 1][sr] = (unsigned short)(w.x >> 16);
        v_lds[d0 + 2][sr] = (unsigned short)(w.y); v_lds[d0 + 3][sr] = (unsigned short)(w.y >> 16);
        v_lds[d0 + 4][sr] = (unsigned short)(w.z); v_lds[d0 + 5][sr] = (unsigned short)(w.z >> 16);
        v_lds[d0 + 6][sr] = (unsigned short)(w.w); v_lds[d0 + 7][sr] = (unsigned short)(w.w >> 16);
      }
    }
#if USE_ASYNC_COPY
    async_wait0();
#endif
    __syncthreads();

    if (kk < kend_wave) {                                 // wave-uniform causal early-out
      // S = Q @ K^T for 2 key sub-tiles of 16
      v8f s[2];
#pragma unroll
      for (int sub = 0; sub < 2; ++sub) {
        Frag bfr[4];
        const int key_l = sub * 16 + l16;
#pragma unroll
        for (int f = 0; f < 4; ++f)
#pragma unroll
          for (int vv = 0; vv < 8; ++vv) {
            int d = f * 32 + 2 * vv + (vv >= 4 ? 8 : 0) + half * 8;
            bfr[f].u[vv] = k_lds[key_l][d >> 1];
          }
        v8f accv;
        zero8(accv);
#pragma unroll
        for (int f = 0; f < 4; ++f) accv = wmma_bf16(af[f], bfr[f], accv);
        s[sub] = accv;
      }

      // causal mask + online softmax
      float rscale[8];
#pragma unroll
      for (int i = 0; i < 8; ++i) {
        const int qrow = qbase + i + half * 8;
        const int k0 = kk + l16, k1 = kk + 16 + l16;
        float s0 = s[0][i] * scale, s1 = s[1][i] * scale;
        if (k0 > qrow) s0 = -1e30f;
        if (k1 > qrow) s1 = -1e30f;
        float mx = half_max(fmaxf(s0, s1));
        float mnew = fmaxf(mrow[i], mx);
        float p0 = __expf(s0 - mnew), p1 = __expf(s1 - mnew);
        s[0][i] = p0; s[1][i] = p1;
        float sum = half_sum(p0 + p1);
        rscale[i] = __expf(mrow[i] - mnew);
        lrow[i] = lrow[i] * rscale[i] + sum;
        mrow[i] = mnew;
      }

      // C-layout P -> per-wave LDS -> A-fragment (wave-local LDS is in-order)
#pragma unroll
      for (int i = 0; i < 8; ++i) {
        const int r = i + half * 8;
        p_lds[wid][r][l16]      = f2bf(s[0][i]);
        p_lds[wid][r][16 + l16] = f2bf(s[1][i]);
      }
      Frag pf;
#pragma unroll
      for (int vv = 0; vv < 8; ++vv) {
        int kb = 2 * vv + (vv >= 4 ? 8 : 0) + half * 8;
        pf.u[vv] = ((const unsigned*)p_lds[wid])[l16 * 16 + (kb >> 1)];
      }
      // O = O*rescale + P @ V
#pragma unroll
      for (int c = 0; c < 8; ++c) {
        Frag bv;
        const int d = c * 16 + l16;
#pragma unroll
        for (int vv = 0; vv < 8; ++vv) {
          int kx = 2 * vv + (vv >= 4 ? 8 : 0) + half * 8;
          bv.u[vv] = *(const unsigned*)&v_lds[d][kx];
        }
        v8f oc = oacc[c];
#pragma unroll
        for (int i = 0; i < 8; ++i) oc[i] *= rscale[i];
        oacc[c] = wmma_bf16(pf, bv, oc);
      }
    }
    __syncthreads();                                      // tile reads done before restage
  }
  // epilogue: normalize and write bf16 O rows
#pragma unroll
  for (int c = 0; c < 8; ++c)
#pragma unroll
    for (int i = 0; i < 8; ++i) {
      const int row = qbase + i + half * 8;
      const int col = c * 16 + l16;
      o[(size_t)row * HID_ + h * HD_ + col] = f2bf(oacc[c][i] / lrow[i]);
    }
}

// ---------------- router logits (tiny GEMV) ----------------
__global__ __launch_bounds__(32) void k_router(const unsigned short* __restrict__ h2,
                                               const float* __restrict__ gw,
                                               float* __restrict__ logits) {
  const int t = blockIdx.x, lane = threadIdx.x;
  const unsigned short* hr = h2 + (size_t)t * HID_;
#pragma unroll
  for (int e = 0; e < NE_; ++e) {
    const float* g = gw + (size_t)e * HID_;
    float s = 0.f;
    for (int d = lane; d < HID_; d += 32) s += bf2f(hr[d]) * g[d];
    s = wave_sum(s);
    if (lane == 0) logits[t * NE_ + e] = s;
  }
}

// ---------------- top-2 routing: build 64-row-padded per-expert token lists ----------------
__device__ inline void top2(const float* lg, int& i1, int& i2, float& v1, float& v2) {
  i1 = 0; v1 = lg[0];
#pragma unroll
  for (int e = 1; e < NE_; ++e) if (lg[e] > v1) { v1 = lg[e]; i1 = e; }
  i2 = (i1 == 0) ? 1 : 0; v2 = lg[i2];
#pragma unroll
  for (int e = 0; e < NE_; ++e) if (e != i1 && lg[e] > v2) { v2 = lg[e]; i2 = e; }
}

__global__ __launch_bounds__(256) void k_route(const float* __restrict__ logits,
                                               int* __restrict__ off_g, int* __restrict__ cnt_g,
                                               int* __restrict__ tok, int* __restrict__ slot,
                                               float* __restrict__ wt) {
  __shared__ int cnt[NE_], fill[NE_], offs[NE_ + 1];
  if (threadIdx.x < NE_) { cnt[threadIdx.x] = 0; fill[threadIdx.x] = 0; }
  __syncthreads();
  for (int t = threadIdx.x; t < TT; t += 256) {
    int i1, i2; float v1, v2;
    top2(logits + t * NE_, i1, i2, v1, v2);
    atomicAdd(&cnt[i1], 1);
    atomicAdd(&cnt[i2], 1);
  }
  __syncthreads();
  if (threadIdx.x == 0) {
    int o = 0;
    for (int e = 0; e < NE_; ++e) { offs[e] = o; o += (cnt[e] + 63) & ~63; }
    offs[NE_] = o;
    for (int e = 0; e <= NE_; ++e) off_g[e] = offs[e];
    for (int e = 0; e < NE_; ++e) cnt_g[e] = cnt[e];
  }
  __syncthreads();
  for (int t = threadIdx.x; t < TT; t += 256) {
    int i1, i2; float v1, v2;
    top2(logits + t * NE_, i1, i2, v1, v2);
    float w1n = 1.0f / (1.0f + __expf(v2 - v1));          // softmax-top2 renormalized
    int p = offs[i1] + atomicAdd(&fill[i1], 1);
    tok[p] = t; slot[p] = 0; wt[p] = w1n;
    p = offs[i2] + atomicAdd(&fill[i2], 1);
    tok[p] = t; slot[p] = 1; wt[p] = 1.0f - w1n;
  }
  __syncthreads();
  for (int p = threadIdx.x; p < offs[NE_]; p += 256) {    // init padding rows
    int e = 0;
#pragma unroll
    for (int i = 0; i < NE_ - 1; ++i) if (offs[i + 1] <= p) e = i + 1;
    if (p - offs[e] >= cnt[e]) { tok[p] = 0; slot[p] = 0; wt[p] = 0.f; }
  }
}

// ---------------- fused w1/w3 expert GEMM: G = silu(h@w1^T) * (h@w3^T) ----------------
__global__ __launch_bounds__(256) void k_gemm_w1w3(const unsigned short* __restrict__ h2,
                                                   const float* __restrict__ w1,
                                                   const float* __restrict__ w3,
                                                   const int* __restrict__ off,
                                                   const int* __restrict__ tok,
                                                   unsigned short* __restrict__ G) {
  const int nbase = blockIdx.x * BN, mbase = blockIdx.y * BM;
  if (mbase >= off[8]) return;                            // device-side dynamic extent
  int e = 0;
#pragma unroll
  for (int i = 0; i < 7; ++i) if (off[i + 1] <= mbase) e = i + 1;
  const float* B0 = w1 + (size_t)e * FFN_ * HID_;
  const float* B1 = w3 + (size_t)e * FFN_ * HID_;
  v8f acc[2][2][2];
#pragma unroll
  for (int nb = 0; nb < 2; ++nb)
#pragma unroll
    for (int i = 0; i < 2; ++i)
#pragma unroll
      for (int j = 0; j < 2; ++j) zero8(acc[nb][i][j]);
  gemm_tile<2>(h2, B0, B1, HID_, mbase, nbase, tok, acc);
  const int lane = threadIdx.x & 31, wid = threadIdx.x >> 5;
  const int wm = wid >> 2, wn = wid & 3, half = lane >> 4, l16 = lane & 15;
#pragma unroll
  for (int i = 0; i < 2; ++i)
#pragma unroll
    for (int j = 0; j < 2; ++j)
#pragma unroll
      for (int vi = 0; vi < 8; ++vi) {
        int row = mbase + wm * 32 + i * 16 + vi + half * 8;
        int col = nbase + wn * 32 + j * 16 + l16;
        float g1 = acc[0][i][j][vi], g3 = acc[1][i][j][vi];
        float sig = 1.0f / (1.0f + __expf(-g1));
        G[(size_t)row * FFN_ + col] = f2bf(g1 * sig * g3);
      }
}

// ---------------- w2 expert GEMM + weighted scatter into per-slot buffers ----------------
__global__ __launch_bounds__(256) void k_gemm_w2(const unsigned short* __restrict__ G,
                                                 const float* __restrict__ w2,
                                                 const int* __restrict__ off,
                                                 const int* __restrict__ cnt,
                                                 const int* __restrict__ tok,
                                                 const int* __restrict__ slot,
                                                 const float* __restrict__ wt,
                                                 float* __restrict__ buf) {
  const int nbase = blockIdx.x * BN, mbase = blockIdx.y * BM;
  if (mbase >= off[8]) return;
  int e = 0;
#pragma unroll
  for (int i = 0; i < 7; ++i) if (off[i + 1] <= mbase) e = i + 1;
  const float* B = w2 + (size_t)e * HID_ * FFN_;
  v8f acc[1][2][2];
#pragma unroll
  for (int i = 0; i < 2; ++i)
#pragma unroll
    for (int j = 0; j < 2; ++j) zero8(acc[0][i][j]);
  gemm_tile<1>(G, B, nullptr, FFN_, mbase, nbase, nullptr, acc);
  const int oe = off[e], ce = cnt[e];
  const int lane = threadIdx.x & 31, wid = threadIdx.x >> 5;
  const int wm = wid >> 2, wn = wid & 3, half = lane >> 4, l16 = lane & 15;
#pragma unroll
  for (int i = 0; i < 2; ++i)
#pragma unroll
    for (int j = 0; j < 2; ++j)
#pragma unroll
      for (int vi = 0; vi < 8; ++vi) {
        int p = mbase + wm * 32 + i * 16 + vi + half * 8;
        if (p - oe < ce) {                                // skip padding rows
          int t = tok[p], s = slot[p];
          int col = nbase + wn * 32 + j * 16 + l16;
          buf[((size_t)s * TT + t) * HID_ + col] = wt[p] * acc[0][i][j][vi];
        }
      }
}

// ---------------- final residual + MoE combine ----------------
__global__ __launch_bounds__(256) void k_final(const float* __restrict__ h2,
                                               const float* __restrict__ buf,
                                               float* __restrict__ out) {
  const size_t i = (size_t)blockIdx.x * 256 + threadIdx.x;
  out[i] = h2[i] + buf[i] + buf[(size_t)TT * HID_ + i];
}

// ---------------- host-side orchestration ----------------
extern "C" void kernel_launch(void* const* d_in, const int* in_sizes, int n_in,
                              void* d_out, int out_size, void* d_ws, size_t ws_size,
                              hipStream_t stream) {
  const float* hidden = (const float*)d_in[0];
  const int*   pos    = (const int*)d_in[1];
  const float* ln1    = (const float*)d_in[2];
  const float* ln2    = (const float*)d_in[3];
  const float* wqkv   = (const float*)d_in[4];
  const float* wo     = (const float*)d_in[5];
  const float* gate   = (const float*)d_in[6];
  const float* w1     = (const float*)d_in[7];
  const float* w2     = (const float*)d_in[8];
  const float* w3     = (const float*)d_in[9];
  float* out = (float*)d_out;

  // workspace layout (~172 MiB total)
  char* W = (char*)d_ws;
  unsigned short* h1  = (unsigned short*)W; W += (size_t)TT * HID_ * 2;        // ln1 output, bf16
  float* qkv          = (float*)W;          W += (size_t)TT * QKV_N * 4;       // qkv f32
  unsigned short* qb  = (unsigned short*)W; W += (size_t)TT * NH_ * HD_ * 2;   // roped Q bf16
  unsigned short* kb  = (unsigned short*)W; W += (size_t)TT * NKV_ * HD_ * 2;  // roped K bf16
  unsigned short* vb  = (unsigned short*)W; W += (size_t)TT * NKV_ * HD_ * 2;  // V bf16
  unsigned short* ob  = (unsigned short*)W; W += (size_t)TT * HID_ * 2;        // attn out bf16
  float* hidden2      = (float*)W;          W += (size_t)TT * HID_ * 4;        // post-attn resid
  unsigned short* h2b = (unsigned short*)W; W += (size_t)TT * HID_ * 2;        // ln2 output bf16
  float* logits       = (float*)W;          W += (size_t)TT * NE_ * 4;
  int* offp           = (int*)W;            W += 64;
  int* cntp           = (int*)W;            W += 64;
  int* tokp           = (int*)W;            W += (size_t)PADCAP * 4;
  int* slotp          = (int*)W;            W += (size_t)PADCAP * 4;
  float* wtp          = (float*)W;          W += (size_t)PADCAP * 4;
  unsigned short* G   = (unsigned short*)W; W += (size_t)PADCAP * FFN_ * 2;    // gated act bf16
  float* moebuf       = (float*)W;          W += (size_t)2 * TT * HID_ * 4;    // per-slot outputs

  k_rmsnorm<<<TT, 256, 0, stream>>>(hidden, ln1, h1);
  k_gemm_qkv<<<dim3(QKV_N / BN, TT / BM), 256, 0, stream>>>(h1, wqkv, qkv);
  k_rope<<<TT, 256, 0, stream>>>(qkv, pos, qb, kb, vb);
  k_attn<<<dim3(NH_, TT / 64), 128, 0, stream>>>(qb, kb, vb, ob);
  k_gemm_wo<<<dim3(HID_ / BN, TT / BM), 256, 0, stream>>>(ob, wo, hidden, hidden2);
  k_rmsnorm<<<TT, 256, 0, stream>>>(hidden2, ln2, h2b);
  k_router<<<TT, 32, 0, stream>>>(h2b, gate, logits);
  k_route<<<1, 256, 0, stream>>>(logits, offp, cntp, tokp, slotp, wtp);
  k_gemm_w1w3<<<dim3(FFN_ / BN, PADCAP / BM), 256, 0, stream>>>(h2b, w1, w3, offp, tokp, G);
  k_gemm_w2<<<dim3(HID_ / BN, PADCAP / BM), 256, 0, stream>>>(G, w2, offp, cntp, tokp, slotp, wtp, moebuf);
  k_final<<<(TT * HID_) / 256, 256, 0, stream>>>(hidden2, moebuf, out);
}
